// Masked_MAB_PT_87162066305551
// MI455X (gfx1250) — matrix operand order repaired and checked by reading, not compiled, not run on hardware
//
#include <hip/hip_runtime.h>
#include <math.h>

// ---------------- types / helpers ----------------
typedef __bf16 bf16_t;
typedef __attribute__((ext_vector_type(16))) bf16_t v16bf;
typedef __attribute__((ext_vector_type(8)))  float  v8f;

#define WMMA_BF16(a, b, c) \
  __builtin_amdgcn_wmma_f32_16x16x32_bf16(false, (a), false, (b), (short)0, (c), false, false)

__device__ __forceinline__ unsigned short f2bf(float f) {
  unsigned int u = __builtin_bit_cast(unsigned int, f);
  u += 0x7FFFu + ((u >> 16) & 1u);          // round-to-nearest-even
  return (unsigned short)(u >> 16);
}

union ABf {
  unsigned short u[16];
  uint4 q[2];       // two 16-byte chunks
  v16bf v;
};

// Fragment striping (per ISA tables):
// A (16x32 bf16): lane row = lane%16; u[0..7] <- K = hs*8 + 0..7 ; u[8..15] <- K = 16 + hs*8 + 0..7
// B (32x16 bf16): lane col = lane%16; u[0..7] <- K = hs*8 + 0..7 ; u[8..15] <- K = 16 + hs*8 + 0..7
// C (16x16 f32):  lane col = lane%16; acc[v]  -> row = v + 8*hs

// Problem constants
#define BB 8
#define NSEQ 1024
#define DMODEL 512
#define HH 8
#define HDIM 64
#define ROWS (BB * NSEQ)   // 8192

// ---------------- prepass: convert activations to bf16 ----------------
__global__ void cvt_act_kernel(const float* __restrict__ Q, const float* __restrict__ K,
                               unsigned short* __restrict__ Qc, unsigned short* __restrict__ Kc) {
  int idx = blockIdx.x * blockDim.x + threadIdx.x;   // NE threads
  Qc[idx] = f2bf(Q[idx]);
  Kc[idx] = f2bf(K[idx]);
}

// ---------------- prepass: convert + transpose weights: Wt[m][c][k] = W[k][c] ------
__global__ void cvt_w_kernel(const float* __restrict__ Wq, const float* __restrict__ Wk,
                             const float* __restrict__ Wv, const float* __restrict__ Wo,
                             unsigned short* __restrict__ Wt) {
  int idx = blockIdx.x * blockDim.x + threadIdx.x;   // 4 * 512 * 512
  int m = idx >> 18;
  int r = idx & 262143;
  int c = r >> 9;
  int k = r & 511;
  const float* W = (m == 0) ? Wq : (m == 1) ? Wk : (m == 2) ? Wv : Wo;
  Wt[idx] = f2bf(W[k * DMODEL + c]);
}

// ---------------- kernel 1: q/k/v projections ----------------
// one wave per 16x16 output tile; 3 GEMMs of (8192x512)@(512x512), bf16 in, f32 acc
__global__ void proj_kernel(const unsigned short* __restrict__ Qc,
                            const unsigned short* __restrict__ Kc,
                            const float* __restrict__ mask,
                            const unsigned short* __restrict__ Wt,
                            const float* __restrict__ bq, const float* __restrict__ bk,
                            const float* __restrict__ bv,
                            float* __restrict__ qf, unsigned short* __restrict__ qb,
                            unsigned short* __restrict__ kbuf, unsigned short* __restrict__ vT) {
  int wid  = (blockIdx.x * blockDim.x + threadIdx.x) >> 5;
  int lane = threadIdx.x & 31;
  int gemm = wid >> 14;          // 16384 tiles per GEMM
  int tile = wid & 16383;
  int tm = tile >> 5;            // 0..511 row tile
  int tn = tile & 31;            // 0..31  col tile
  const unsigned short* src = (gemm == 0) ? Qc : Kc;
  const float* bias = (gemm == 0) ? bq : (gemm == 1) ? bk : bv;
  const unsigned short* wmat = Wt + (size_t)gemm * DMODEL * DMODEL;

  int hs = lane >> 4, lc = lane & 15;
  v8f acc = {0.f, 0.f, 0.f, 0.f, 0.f, 0.f, 0.f, 0.f};
  const unsigned short* arow = src + (size_t)(tm * 16 + lc) * DMODEL;
  const unsigned short* brow = wmat + (size_t)(tn * 16 + lc) * DMODEL;

  for (int kb = 0; kb < 16; ++kb) {
    int kbase = kb * 32;
    ABf a, b;
    a.q[0] = *(const uint4*)(arow + kbase + hs * 8);
    a.q[1] = *(const uint4*)(arow + kbase + 16 + hs * 8);
    b.q[0] = *(const uint4*)(brow + kbase + hs * 8);
    b.q[1] = *(const uint4*)(brow + kbase + 16 + hs * 8);
    acc = WMMA_BF16(a.v, b.v, acc);
  }
  int col = tn * 16 + lc;
  float bcol = bias[col];
#pragma unroll
  for (int v = 0; v < 8; ++v) {
    int row = tm * 16 + v + 8 * hs;
    float val = (acc[v] + bcol) * mask[row];
    unsigned short vb16 = f2bf(val);
    if (gemm == 0) {
      size_t idx = (size_t)row * DMODEL + col;
      qf[idx] = val;
      qb[idx] = vb16;
    } else if (gemm == 1) {
      kbuf[(size_t)row * DMODEL + col] = vb16;
    } else {
      // transposed V: vT[((b*H + h)*HDIM + d)*NSEQ + i]
      int b_ = row >> 10, i = row & 1023;
      int h = col >> 6, dd = col & 63;
      vT[((size_t)(b_ * HH + h) * HDIM + dd) * NSEQ + i] = vb16;
    }
  }
}

// ---------------- kernel 2: attention (scores + masked softmax + A@V + residual) ----
// grid: (N/16)*H*B blocks (b fastest for U L2 locality); wave w owns j in [w*128,w*128+128)
__global__ void attn_kernel(const float* __restrict__ U, const float* __restrict__ mask,
                            const float* __restrict__ qf,
                            const unsigned short* __restrict__ qb,
                            const unsigned short* __restrict__ kbuf,
                            const unsigned short* __restrict__ vT,
                            float* __restrict__ Of, unsigned short* __restrict__ Ob) {
  __shared__ __align__(16) unsigned short e_s[16 * NSEQ];  // 32KB bf16 probs; reused as f32 o_red
  __shared__ float redbuf[16][8];
  __shared__ float rowmax[16];
  __shared__ float rowinv[16];
  __shared__ float mrow[16];

  int blk = blockIdx.x;
  int b = blk & 7;            // fastest: adjacent blocks share U cachelines
  int h = (blk >> 3) & 7;
  int i0 = (blk >> 6) * 16;
  int w = threadIdx.x >> 5;
  int lane = threadIdx.x & 31;
  int hs = lane >> 4, lc = lane & 15;

  if (threadIdx.x < 16) mrow[threadIdx.x] = mask[b * NSEQ + i0 + threadIdx.x];
  __syncthreads();

  const float scale = 0.044194173824159216f;  // 1/sqrt(512)

  // qh A-fragments (shared across all 8 column tiles), 2 k-steps over DH=64
  ABf aq[2];
  {
    const unsigned short* qrow = qb + (size_t)(b * NSEQ + i0 + lc) * DMODEL + h * HDIM;
#pragma unroll
    for (int kb = 0; kb < 2; ++kb) {
      aq[kb].q[0] = *(const uint4*)(qrow + kb * 32 + hs * 8);
      aq[kb].q[1] = *(const uint4*)(qrow + kb * 32 + 16 + hs * 8);
    }
  }

  // scores for this wave's 16x128 slab, kept in C-layout registers
  v8f sc[8];
#pragma unroll
  for (int ct = 0; ct < 8; ++ct) {
    int j0 = w * 128 + ct * 16;
    v8f acc = {0.f, 0.f, 0.f, 0.f, 0.f, 0.f, 0.f, 0.f};
    const unsigned short* kr = kbuf + (size_t)(b * NSEQ + j0 + lc) * DMODEL + h * HDIM;
#pragma unroll
    for (int kb = 0; kb < 2; ++kb) {
      ABf bfr;
      bfr.q[0] = *(const uint4*)(kr + kb * 32 + hs * 8);
      bfr.q[1] = *(const uint4*)(kr + kb * 32 + 16 + hs * 8);
      acc = WMMA_BF16(aq[kb].v, bfr.v, acc);
    }
    // + bias U[h,i,j,b]; prefetch next tile's U stream into cache
    size_t ub = (((size_t)h * NSEQ + (i0 + 8 * hs)) * NSEQ + (j0 + lc)) * BB + b;
    if (ct < 7)
      __builtin_prefetch(U + ub + 16 * BB, 0, 1);   // next j0 (+16 cols), same row
#pragma unroll
    for (int v = 0; v < 8; ++v) {
      float bias = U[ub + (size_t)v * NSEQ * BB];
      acc[v] = acc[v] * scale + bias;
    }
    sc[ct] = acc;
  }

  // row max: per-lane over 8 col-tiles, shfl across the 16-lane half, LDS across waves
#pragma unroll
  for (int v = 0; v < 8; ++v) {
    float m = sc[0][v];
#pragma unroll
    for (int ct = 1; ct < 8; ++ct) m = fmaxf(m, sc[ct][v]);
#pragma unroll
    for (int off = 1; off < 16; off <<= 1) m = fmaxf(m, __shfl_xor(m, off, 32));
    if (lc == 0) redbuf[v + 8 * hs][w] = m;
  }
  __syncthreads();
  if (threadIdx.x < 16) {
    float m = redbuf[threadIdx.x][0];
    for (int ww = 1; ww < 8; ++ww) m = fmaxf(m, redbuf[threadIdx.x][ww]);
    rowmax[threadIdx.x] = m;
  }
  __syncthreads();

  // masked exp: num = exp((s-max)*mi*mj)*mi*mj ; store UNNORMALIZED bf16 probs to LDS
  float psum[8];
#pragma unroll
  for (int v = 0; v < 8; ++v) psum[v] = 0.f;
#pragma unroll
  for (int ct = 0; ct < 8; ++ct) {
    int j = w * 128 + ct * 16 + lc;
    float mj = mask[b * NSEQ + j];
#pragma unroll
    for (int v = 0; v < 8; ++v) {
      int r = v + 8 * hs;
      float mm = mrow[r] * mj;
      float e = __expf((sc[ct][v] - rowmax[r]) * mm) * mm;
      psum[v] += e;
      e_s[r * NSEQ + j] = f2bf(e);
    }
  }
#pragma unroll
  for (int v = 0; v < 8; ++v) {
    float s = psum[v];
#pragma unroll
    for (int off = 1; off < 16; off <<= 1) s += __shfl_xor(s, off, 32);
    if (lc == 0) redbuf[v + 8 * hs][w] = s;
  }
  __syncthreads();
  if (threadIdx.x < 16) {
    float s = 0.f;
    for (int ww = 0; ww < 8; ++ww) s += redbuf[threadIdx.x][ww];
    rowinv[threadIdx.x] = 1.0f / (s + 1e-16f);
  }
  __syncthreads();

  // num@V over this wave's 128 j's: 4 k-steps x 4 d-tiles (row-normalization deferred)
  v8f zero = {0.f, 0.f, 0.f, 0.f, 0.f, 0.f, 0.f, 0.f};
  v8f oacc[4] = {zero, zero, zero, zero};
#pragma unroll
  for (int ks = 0; ks < 4; ++ks) {
    int jb = w * 128 + ks * 32;
    ABf afr;
    afr.q[0] = *(const uint4*)(e_s + lc * NSEQ + jb + hs * 8);        // ds_load_b128
    afr.q[1] = *(const uint4*)(e_s + lc * NSEQ + jb + 16 + hs * 8);
#pragma unroll
    for (int ct = 0; ct < 4; ++ct) {
      ABf bfr;
      const unsigned short* vr =
          vT + ((size_t)(b * HH + h) * HDIM + ct * 16 + lc) * NSEQ + jb;
      bfr.q[0] = *(const uint4*)(vr + hs * 8);
      bfr.q[1] = *(const uint4*)(vr + 16 + hs * 8);
      oacc[ct] = WMMA_BF16(afr.v, bfr.v, oacc[ct]);
    }
  }
  __syncthreads();  // all prob reads done -> reuse e_s as float o_red[8][16][64]
  float* o_red = reinterpret_cast<float*>(e_s);
#pragma unroll
  for (int ct = 0; ct < 4; ++ct)
#pragma unroll
    for (int v = 0; v < 8; ++v)
      o_red[(w * 16 + v + 8 * hs) * 64 + ct * 16 + lc] = oacc[ct][v];
  __syncthreads();

  // cross-wave reduce, apply 1/rowsum, add residual qh, store f32 and bf16 O
  for (int pos = threadIdx.x; pos < 16 * 64; pos += blockDim.x) {
    int r = pos >> 6, d = pos & 63;
    float s = 0.f;
#pragma unroll
    for (int ww = 0; ww < 8; ++ww) s += o_red[(ww * 16 + r) * 64 + d];
    size_t gi = (size_t)(b * NSEQ + i0 + r) * DMODEL + h * HDIM + d;
    float val = qf[gi] + s * rowinv[r];
    Of[gi] = val;
    Ob[gi] = f2bf(val);
  }
}

// ---------------- kernel 3: out = O + relu((O@Wo + bo) * m) ----------------
__global__ void outproj_kernel(const unsigned short* __restrict__ Wt,  // matrix 3
                               const float* __restrict__ bo,
                               const float* __restrict__ mask,
                               const float* __restrict__ Of,
                               const unsigned short* __restrict__ Ob,
                               float* __restrict__ out) {
  int wid  = (blockIdx.x * blockDim.x + threadIdx.x) >> 5;
  int lane = threadIdx.x & 31;
  int tm = wid >> 5;
  int tn = wid & 31;
  int hs = lane >> 4, lc = lane & 15;
  v8f acc = {0.f, 0.f, 0.f, 0.f, 0.f, 0.f, 0.f, 0.f};
  const unsigned short* arow = Ob + (size_t)(tm * 16 + lc) * DMODEL;
  const unsigned short* brow = Wt + (size_t)3 * DMODEL * DMODEL + (size_t)(tn * 16 + lc) * DMODEL;
  for (int kb = 0; kb < 16; ++kb) {
    int kbase = kb * 32;
    ABf a, b;
    a.q[0] = *(const uint4*)(arow + kbase + hs * 8);
    a.q[1] = *(const uint4*)(arow + kbase + 16 + hs * 8);
    b.q[0] = *(const uint4*)(brow + kbase + hs * 8);
    b.q[1] = *(const uint4*)(brow + kbase + 16 + hs * 8);
    acc = WMMA_BF16(a.v, b.v, acc);
  }
  int col = tn * 16 + lc;
  float bcol = bo[col];
#pragma unroll
  for (int v = 0; v < 8; ++v) {
    int row = tm * 16 + v + 8 * hs;
    float t = (acc[v] + bcol) * mask[row];
    t = fmaxf(t, 0.f);
    size_t idx = (size_t)row * DMODEL + col;
    out[idx] = Of[idx] + t;
  }
}

// ---------------- launch ----------------
extern "C" void kernel_launch(void* const* d_in, const int* in_sizes, int n_in,
                              void* d_out, int out_size, void* d_ws, size_t ws_size,
                              hipStream_t stream) {
  const float* Q    = (const float*)d_in[0];
  const float* K    = (const float*)d_in[1];
  const float* U    = (const float*)d_in[2];
  const float* mask = (const float*)d_in[3];
  const float* Wq = (const float*)d_in[4];
  const float* bq = (const float*)d_in[5];
  const float* Wk = (const float*)d_in[6];
  const float* bk = (const float*)d_in[7];
  const float* Wv = (const float*)d_in[8];
  const float* bv = (const float*)d_in[9];
  const float* Wo = (const float*)d_in[10];
  const float* bo = (const float*)d_in[11];
  float* out = (float*)d_out;

  char* ws = (char*)d_ws;
  const size_t NE = (size_t)ROWS * DMODEL;  // 8192*512 = 4M
  float* qf = (float*)ws;                     ws += NE * sizeof(float);
  float* Of = (float*)ws;                     ws += NE * sizeof(float);
  unsigned short* qb   = (unsigned short*)ws; ws += NE * 2;
  unsigned short* kbuf = (unsigned short*)ws; ws += NE * 2;
  unsigned short* vT   = (unsigned short*)ws; ws += NE * 2;
  unsigned short* Ob   = (unsigned short*)ws; ws += NE * 2;
  unsigned short* Qc   = (unsigned short*)ws; ws += NE * 2;
  unsigned short* Kc   = (unsigned short*)ws; ws += NE * 2;
  unsigned short* Wt   = (unsigned short*)ws; ws += (size_t)4 * DMODEL * DMODEL * 2;

  cvt_act_kernel<<<NE / 256, 256, 0, stream>>>(Q, K, Qc, Kc);
  cvt_w_kernel<<<(4 * DMODEL * DMODEL) / 256, 256, 0, stream>>>(Wq, Wk, Wv, Wo, Wt);
  // 3 GEMMs x 16384 wave-tiles / 8 waves per block
  proj_kernel<<<6144, 256, 0, stream>>>(Qc, Kc, mask, Wt, bq, bk, bv, qf, qb, kbuf, vT);
  // (N/16)*H*B = 4096 blocks, b fastest for U locality
  attn_kernel<<<4096, 256, 0, stream>>>(U, mask, qf, qb, kbuf, vT, Of, Ob);
  // 16384 wave-tiles / 8
  outproj_kernel<<<2048, 256, 0, stream>>>(Wt, bo, mask, Of, Ob, out);
}
